// ToggleGNNv2_90855738180235
// MI455X (gfx1250) — compile-verified, matrix operand order checked by reading, not run on hardware
//
#include <hip/hip_runtime.h>
#include <hip/hip_bf16.h>

#define N_NODES 100000
#define N_EDGES 3200000
#define IN_CH   35
#define NUM_FAM 26
#define EMBED   8
#define NEW_IN  17      // 6 + 8 + 3
#define K1PAD   32      // layer-1 K padded to 32 for WMMA
#define HID     64
#define LN_EPS  1e-5f

typedef __attribute__((ext_vector_type(16))) _Float16 v16h;
typedef __attribute__((ext_vector_type(8)))  float    v8f;

// ---------------------------------------------------------------------------
// Node feature preprocessing: embed(onehot) -> LayerNorm -> h0 (stride K1PAD)
// ---------------------------------------------------------------------------
__global__ void embed_ln_kernel(const float* __restrict__ x,
                                const float* __restrict__ Wemb,   // 26 x 8
                                const float* __restrict__ bemb,   // 8
                                const float* __restrict__ ln_g,   // 8
                                const float* __restrict__ ln_b,   // 8
                                float* __restrict__ h0)           // N x K1PAD
{
    int i = blockIdx.x * blockDim.x + threadIdx.x;
    if (i >= N_NODES) return;
    const float* xr = x + (size_t)i * IN_CH;

    float emb[EMBED];
    #pragma unroll
    for (int j = 0; j < EMBED; ++j) {
        float s = bemb[j];
        #pragma unroll
        for (int f = 0; f < NUM_FAM; ++f)
            s += xr[6 + f] * Wemb[f * EMBED + j];
        emb[j] = s;
    }
    float mu = 0.f;
    #pragma unroll
    for (int j = 0; j < EMBED; ++j) mu += emb[j];
    mu *= (1.f / EMBED);
    float var = 0.f;
    #pragma unroll
    for (int j = 0; j < EMBED; ++j) { float d = emb[j] - mu; var += d * d; }
    var *= (1.f / EMBED);
    float rs = __frsqrt_rn(var + LN_EPS);

    float* o = h0 + (size_t)i * K1PAD;
    #pragma unroll
    for (int c = 0; c < 6; ++c) o[c] = xr[c];
    #pragma unroll
    for (int j = 0; j < EMBED; ++j)
        o[6 + j] = (emb[j] - mu) * rs * ln_g[j] + ln_b[j];
    #pragma unroll
    for (int c = 0; c < 3; ++c) o[14 + c] = xr[32 + c];
    #pragma unroll
    for (int c = NEW_IN; c < K1PAD; ++c) o[c] = 0.f;   // zero pad
}

// ---------------------------------------------------------------------------
// In-degree count (computed once, reused for all 3 layers)
// ---------------------------------------------------------------------------
__global__ void degree_kernel(const int* __restrict__ eidx, float* __restrict__ cnt)
{
    int e = blockIdx.x * blockDim.x + threadIdx.x;
    if (e >= N_EDGES) return;
    atomicAdd(&cnt[eidx[N_EDGES + e]], 1.0f);
}

__global__ void invcnt_kernel(const float* __restrict__ cnt, float* __restrict__ inv)
{
    int i = blockIdx.x * blockDim.x + threadIdx.x;
    if (i >= N_NODES) return;
    inv[i] = 1.0f / fmaxf(cnt[i], 1.0f);
}

// ---------------------------------------------------------------------------
// Pre-pack weight matrix into wave32 WMMA B-fragment layout (f16, K padded).
// Wp index = (((k0/32)*64 + col)*2 + half)*16 + hh
// with k = (k0/32)*32 + half*8 + (hh>>3)*16 + (hh&7)   (ISA 7.12.2)
// ---------------------------------------------------------------------------
__global__ void pack_w_kernel(const float* __restrict__ W,    // kActual x 64
                              _Float16* __restrict__ Wp,
                              int kdim, int kActual)
{
    int t = blockIdx.x * blockDim.x + threadIdx.x;
    if (t >= kdim * HID) return;             // (KDIM/32)*64*2*16 == KDIM*64
    int hh   = t & 15;
    int half = (t >> 4) & 1;
    int col  = (t >> 5) & 63;
    int tile = t >> 11;
    int k = tile * 32 + half * 8 + ((hh >> 3) << 4) + (hh & 7);
    float v = (k < kActual) ? W[(size_t)k * HID + col] : 0.f;
    Wp[t] = (_Float16)v;
}

// ---------------------------------------------------------------------------
// Edge scatter: msum[dst,:] += h[src,:]
// One thread = (edge, 4 channels): float4 gather + 4 f32 atomics.
// ---------------------------------------------------------------------------
template <int CH, int LOG2G>                  // G = CH/4 channel groups
__global__ void scatter_kernel(const int* __restrict__ eidx,
                               const float* __restrict__ h,
                               float* __restrict__ msum)
{
    long long t = (long long)blockIdx.x * blockDim.x + threadIdx.x;
    if (t >= (long long)N_EDGES * (CH / 4)) return;
    int g  = (int)(t & ((CH / 4) - 1));
    int e  = (int)(t >> LOG2G);
    int c4 = g * 4;
    int src = eidx[e];
    int dst = eidx[N_EDGES + e];
    const float4 v = *(const float4*)(h + (size_t)src * CH + c4);
    float* m = msum + (size_t)dst * CH + c4;
    atomicAdd(m + 0, v.x);
    atomicAdd(m + 1, v.y);
    atomicAdd(m + 2, v.z);
    atomicAdd(m + 3, v.w);
}

// ---------------------------------------------------------------------------
// Dense SAGE update with WMMA:
//   Hout = relu( (msum*inv) @ Wl + bl + Hin @ Wr  [+ Hin] )
// Block = 4 waves (128 thr) -> 16 rows x 64 cols; wave w owns cols [16w,16w+16).
// A/H tiles staged in LDS once per block; B fragments loaded pre-packed (f16).
// ---------------------------------------------------------------------------
template <int KDIM, bool RESIDUAL>
__global__ void sage_wmma_kernel(const float* __restrict__ Ain,   // msum, stride KDIM
                                 const float* __restrict__ inv,   // per-row 1/deg
                                 const float* __restrict__ Hin,   // stride KDIM
                                 const _Float16* __restrict__ Wlp, // packed B frags
                                 const float* __restrict__ bl,     // 64
                                 const _Float16* __restrict__ Wrp, // packed B frags
                                 float* __restrict__ Hout)         // stride HID
{
    __shared__ float sA[16 * KDIM];
    __shared__ float sH[16 * KDIM];

    const int tid  = threadIdx.x;
    const int wave = tid >> 5;               // 0..3 -> col tile
    const int lane = tid & 31;
    const int half = lane >> 4;              // lane half selects K sub-block
    const int l16  = lane & 15;
    const int rowTile = blockIdx.x * 16;

    // --- cooperative stage of contiguous 16 x KDIM tiles into LDS ---
    {
        const float4* gA = (const float4*)(Ain + (size_t)rowTile * KDIM);
        const float4* gH = (const float4*)(Hin + (size_t)rowTile * KDIM);
        float4* lA = (float4*)sA;
        float4* lH = (float4*)sH;
        #pragma unroll
        for (int idx = tid; idx < 16 * KDIM / 4; idx += 128) {
            lA[idx] = gA[idx];
            lH[idx] = gH[idx];
        }
    }
    __syncthreads();

    const int arow = rowTile + l16;          // A row this lane feeds
    const int col  = wave * 16 + l16;        // output column this lane owns
    const float ic = inv[arow];

    v8f acc;
    const float bias = bl[col];
    #pragma unroll
    for (int r = 0; r < 8; ++r) acc[r] = bias;

    #pragma unroll
    for (int k0 = 0; k0 < KDIM; k0 += 32) {
        v16h aM, aH;
        #pragma unroll
        for (int hh = 0; hh < 16; ++hh) {
            const int k = k0 + half * 8 + (hh >> 3) * 16 + (hh & 7);
            const float a = sA[l16 * KDIM + k];
            aM[hh] = (_Float16)(a * ic);
            aH[hh] = (_Float16)(sH[l16 * KDIM + k]);
        }
        const size_t boff = (((size_t)(k0 >> 5) * HID + col) * 2 + half) * 16;
        const v16h bL = *(const v16h*)(Wlp + boff);
        const v16h bR = *(const v16h*)(Wrp + boff);
        acc = __builtin_amdgcn_wmma_f32_16x16x32_f16(
                  false, aM, false, bL, (short)0, acc, false, false);
        acc = __builtin_amdgcn_wmma_f32_16x16x32_f16(
                  false, aH, false, bR, (short)0, acc, false, false);
    }

    // C/D layout: VGPR r holds row (r + 8*half), col = l16 (+ wave*16)
    #pragma unroll
    for (int r = 0; r < 8; ++r) {
        const int orow = half * 8 + r;                 // row within tile
        float v = acc[r];
        if (RESIDUAL) v += sH[orow * KDIM + col];      // KDIM==HID here
        v = fmaxf(v, 0.f);
        Hout[(size_t)(rowTile + orow) * HID + col] = v;
    }
}

// ---------------------------------------------------------------------------
// Final projection: out[i] = h3[i,:] @ Wfc + bfc
// ---------------------------------------------------------------------------
__global__ void fc_kernel(const float* __restrict__ h3,
                          const float* __restrict__ Wfc,
                          const float* __restrict__ bfc,
                          float* __restrict__ out)
{
    int i = blockIdx.x * blockDim.x + threadIdx.x;
    if (i >= N_NODES) return;
    float s = bfc[0];
    const float4* hr = (const float4*)(h3 + (size_t)i * HID);
    const float4* wr = (const float4*)Wfc;
    #pragma unroll
    for (int c = 0; c < HID / 4; ++c) {
        float4 a = hr[c], b = wr[c];
        s += a.x * b.x + a.y * b.y + a.z * b.z + a.w * b.w;
    }
    out[i] = s;
}

// ---------------------------------------------------------------------------
extern "C" void kernel_launch(void* const* d_in, const int* in_sizes, int n_in,
                              void* d_out, int out_size, void* d_ws, size_t ws_size,
                              hipStream_t stream)
{
    const float* x     = (const float*)d_in[0];
    const int*   eidx  = (const int*)  d_in[1];   // (2, N_EDGES)
    const float* Wemb  = (const float*)d_in[2];
    const float* bemb  = (const float*)d_in[3];
    const float* ln_g  = (const float*)d_in[4];
    const float* ln_b  = (const float*)d_in[5];
    const float* Wl1   = (const float*)d_in[6];
    const float* bl1   = (const float*)d_in[7];
    const float* Wr1   = (const float*)d_in[8];
    const float* Wl2   = (const float*)d_in[9];
    const float* bl2   = (const float*)d_in[10];
    const float* Wr2   = (const float*)d_in[11];
    const float* Wl3   = (const float*)d_in[12];
    const float* bl3   = (const float*)d_in[13];
    const float* Wr3   = (const float*)d_in[14];
    const float* Wfc   = (const float*)d_in[15];
    const float* bfc   = (const float*)d_in[16];
    float* out = (float*)d_out;

    // workspace layout (floats)
    float* ws   = (float*)d_ws;
    float* h0   = ws;                               // N x 32
    float* hA   = h0 + (size_t)N_NODES * K1PAD;     // N x 64  (h1, then h3)
    float* hB   = hA + (size_t)N_NODES * HID;       // N x 64  (h2)
    float* msum = hB + (size_t)N_NODES * HID;       // N x 64
    float* cnt  = msum + (size_t)N_NODES * HID;     // N
    float* inv  = cnt + N_NODES;                    // N
    // packed f16 weight fragments (sizes are KDIM*64 halfs each)
    _Float16* wp   = (_Float16*)(inv + N_NODES);
    _Float16* Wl1p = wp;                  // 32*64 = 2048
    _Float16* Wr1p = Wl1p + 2048;
    _Float16* Wl2p = Wr1p + 2048;         // 64*64 = 4096
    _Float16* Wr2p = Wl2p + 4096;
    _Float16* Wl3p = Wr2p + 4096;
    _Float16* Wr3p = Wl3p + 4096;

    const int T = 256;
    const int nodeBlocks = (N_NODES + T - 1) / T;
    const int edgeBlocks = (N_EDGES + T - 1) / T;
    const int tileBlocks = N_NODES / 16;            // 6250, exact

    // --- preprocessing, degree, weight packing ---
    embed_ln_kernel<<<nodeBlocks, T, 0, stream>>>(x, Wemb, bemb, ln_g, ln_b, h0);
    hipMemsetAsync(cnt, 0, (size_t)N_NODES * sizeof(float), stream);
    degree_kernel<<<edgeBlocks, T, 0, stream>>>(eidx, cnt);
    invcnt_kernel<<<nodeBlocks, T, 0, stream>>>(cnt, inv);

    pack_w_kernel<<<(K1PAD * HID + T - 1) / T, T, 0, stream>>>(Wl1, Wl1p, K1PAD, NEW_IN);
    pack_w_kernel<<<(K1PAD * HID + T - 1) / T, T, 0, stream>>>(Wr1, Wr1p, K1PAD, NEW_IN);
    pack_w_kernel<<<(HID * HID + T - 1) / T, T, 0, stream>>>(Wl2, Wl2p, HID, HID);
    pack_w_kernel<<<(HID * HID + T - 1) / T, T, 0, stream>>>(Wr2, Wr2p, HID, HID);
    pack_w_kernel<<<(HID * HID + T - 1) / T, T, 0, stream>>>(Wl3, Wl3p, HID, HID);
    pack_w_kernel<<<(HID * HID + T - 1) / T, T, 0, stream>>>(Wr3, Wr3p, HID, HID);

    // ---- layer 1 (K padded 17 -> 32) ----
    hipMemsetAsync(msum, 0, (size_t)N_NODES * K1PAD * sizeof(float), stream);
    {
        long long tot = (long long)N_EDGES * (K1PAD / 4);
        scatter_kernel<K1PAD, 3><<<(unsigned)((tot + T - 1) / T), T, 0, stream>>>(eidx, h0, msum);
    }
    sage_wmma_kernel<K1PAD, false><<<tileBlocks, 128, 0, stream>>>(
        msum, inv, h0, Wl1p, bl1, Wr1p, hA);

    // ---- layer 2 ----
    hipMemsetAsync(msum, 0, (size_t)N_NODES * HID * sizeof(float), stream);
    {
        long long tot = (long long)N_EDGES * (HID / 4);
        scatter_kernel<HID, 4><<<(unsigned)((tot + T - 1) / T), T, 0, stream>>>(eidx, hA, msum);
    }
    sage_wmma_kernel<HID, true><<<tileBlocks, 128, 0, stream>>>(
        msum, inv, hA, Wl2p, bl2, Wr2p, hB);

    // ---- layer 3 (h3 reuses hA) ----
    hipMemsetAsync(msum, 0, (size_t)N_NODES * HID * sizeof(float), stream);
    {
        long long tot = (long long)N_EDGES * (HID / 4);
        scatter_kernel<HID, 4><<<(unsigned)((tot + T - 1) / T), T, 0, stream>>>(eidx, hB, msum);
    }
    sage_wmma_kernel<HID, true><<<tileBlocks, 128, 0, stream>>>(
        msum, inv, hB, Wl3p, bl3, Wr3p, hA);

    // ---- final projection ----
    fc_kernel<<<nodeBlocks, T, 0, stream>>>(hA, Wfc, bfc, out);
}
